// GCN_83116207112814
// MI455X (gfx1250) — matrix operand order repaired and checked by reading, not compiled, not run on hardware
//
#include <hip/hip_runtime.h>
#include <math.h>

typedef __attribute__((ext_vector_type(16))) _Float16 v16h;
typedef __attribute__((ext_vector_type(8)))  float    v8f;
typedef __attribute__((ext_vector_type(4)))  _Float16 h4;
typedef __attribute__((ext_vector_type(4)))  int      v4i;

#if defined(__has_builtin)
#if __has_builtin(__builtin_amdgcn_global_load_async_to_lds_b128)
#define HAVE_ASYNC_LDS 1
#endif
#if __has_builtin(__builtin_amdgcn_s_wait_asynccnt)
#define HAVE_WAIT_ASYNC 1
#endif
#endif

// ---------------- small utility kernels ----------------

__global__ void k_fill(float* p, float v, int n) {
    int i = blockIdx.x * 256 + threadIdx.x;
    if (i < n) p[i] = v;
}

__global__ void k_deg_edges(const int* __restrict__ dst, float* __restrict__ deg, int E) {
    int e = blockIdx.x * 256 + threadIdx.x;
    if (e < E) atomicAdd(&deg[dst[e]], 1.0f);
}

__global__ void k_rsqrt_inplace(float* d, int N) {
    int i = blockIdx.x * 256 + threadIdx.x;
    if (i < N) d[i] = rsqrtf(fmaxf(d[i], 1.0f));
}

// convert input x (f32, N x 128) -> f16 padded (Npad x 128), zero tail rows
__global__ void k_conv_in(const float* __restrict__ x, _Float16* __restrict__ A16,
                          int N, int Npad) {
    long long idx = (long long)blockIdx.x * 256 + threadIdx.x;   // one float4 per thread
    long long row = idx >> 5;
    if (row >= Npad) return;
    h4 o;
    if (row < N) {
        float4 v = ((const float4*)x)[idx];
        o = h4{(_Float16)v.x, (_Float16)v.y, (_Float16)v.z, (_Float16)v.w};
    } else {
        o = h4{(_Float16)0.f, (_Float16)0.f, (_Float16)0.f, (_Float16)0.f};
    }
    *(h4*)(A16 + idx * 4) = o;
}

// transpose+convert weight: Wt[n*128+k] = (f16) W[k*128+n]
__global__ void k_conv_w(const float* __restrict__ W, _Float16* __restrict__ Wt) {
    int t = blockIdx.x * 256 + threadIdx.x;   // 16384
    int k = t >> 7, n = t & 127;
    Wt[n * 128 + k] = (_Float16)W[k * 128 + n];
}

// one wave per edge: S[dst] += hW[src] * dinv[src]*dinv[dst]
__global__ __launch_bounds__(256)
void k_edge_scatter(const int* __restrict__ srcI, const int* __restrict__ dstI,
                    const float* __restrict__ dinv, const float* __restrict__ hW,
                    float* __restrict__ S, int E) {
    int e = blockIdx.x * 8 + (threadIdx.x >> 5);
    if (e >= E) return;
    int lane = threadIdx.x & 31;
    int s = srcI[e], d = dstI[e];
    float norm = dinv[s] * dinv[d];
    float4 v = ((const float4*)(hW + (long long)s * 128))[lane];
    float* out = S + (long long)d * 128 + lane * 4;
    atomicAdd(out + 0, v.x * norm);
    atomicAdd(out + 1, v.y * norm);
    atomicAdd(out + 2, v.z * norm);
    atomicAdd(out + 3, v.w * norm);
}

// next-layer input: A16 = relu(S + bias) as f16, zero tail rows
__global__ void k_finish_relu(const float* __restrict__ S, const float* __restrict__ bias,
                              _Float16* __restrict__ A16, int N, int Npad) {
    long long idx = (long long)blockIdx.x * 256 + threadIdx.x;
    long long row = idx >> 5;
    if (row >= Npad) return;
    h4 o;
    if (row < N) {
        float4 v  = ((const float4*)S)[idx];
        float4 bb = ((const float4*)bias)[idx & 31];
        o = h4{(_Float16)fmaxf(v.x + bb.x, 0.f), (_Float16)fmaxf(v.y + bb.y, 0.f),
               (_Float16)fmaxf(v.z + bb.z, 0.f), (_Float16)fmaxf(v.w + bb.w, 0.f)};
    } else {
        o = h4{(_Float16)0.f, (_Float16)0.f, (_Float16)0.f, (_Float16)0.f};
    }
    *(h4*)(A16 + idx * 4) = o;
}

// ---------------- WMMA GEMM: C[Npad x 128] = A16 @ W, S = C * dinv^2 ----------------
// Wt holds W transposed (Wt[n][k]) in f16. 8 waves/block, 128 rows/block.
__global__ __launch_bounds__(256)
void k_gemm_wmma(const _Float16* __restrict__ A, const _Float16* __restrict__ Wt,
                 float* __restrict__ C, float* __restrict__ S,
                 const float* __restrict__ dinv, int N) {
    __shared__ _Float16 sB[128 * 136];   // padded stride 136 halves (conflict-free b128 reads)

    const int tid = threadIdx.x;
    // Stage Wt into LDS. Prefer CDNA5 async global->LDS (compiler-visible builtin so
    // the LDS writes are modeled); fall back to plain copy otherwise.
    {
        const float4* gsrc = (const float4*)Wt;
#ifdef HAVE_ASYNC_LDS
        for (int i = tid; i < 2048; i += 256) {
            int row = i >> 4, chunk = i & 15;
            _Float16* dstp = sB + (row * 136 + chunk * 8);
            __builtin_amdgcn_global_load_async_to_lds_b128(
                (__attribute__((address_space(1))) v4i*)(uintptr_t)(gsrc + i),
                (__attribute__((address_space(3))) v4i*)(unsigned)(uintptr_t)dstp,
                0, 0);
        }
#ifdef HAVE_WAIT_ASYNC
        __builtin_amdgcn_s_wait_asynccnt(0);
#else
        asm volatile("s_wait_asynccnt 0x0" ::: "memory");
#endif
#else
        for (int i = tid; i < 2048; i += 256) {
            int row = i >> 4, chunk = i & 15;
            ((float4*)sB)[row * 17 + chunk] = gsrc[i];
        }
#endif
        __syncthreads();
    }

    const int wave = tid >> 5;
    const int lane = tid & 31;
    const int m    = lane & 15;
    const int hi   = lane >> 4;                 // 0 or 1
    const long long rowBase = (long long)blockIdx.x * 128 + wave * 16;

    v8f acc[8];
#pragma unroll
    for (int nt = 0; nt < 8; ++nt) acc[nt] = (v8f)(0.0f);

    union Frag { v16h v; float4 f[2]; };

    const _Float16* arow = A + (rowBase + m) * 128;

#pragma unroll
    for (int ks = 0; ks < 4; ++ks) {
        const int kb = ks * 32;
        // A fragment: lanes 0-15 hold K {kb+0..7, kb+16..23}; lanes 16-31: +8
        Frag a;
        const int k0 = kb + hi * 8;
        a.f[0] = *(const float4*)(arow + k0);
        a.f[1] = *(const float4*)(arow + k0 + 16);
        // batch all 8 B fragments for this k-step, then issue 8 WMMAs
        Frag bf[8];
#pragma unroll
        for (int nt = 0; nt < 8; ++nt) {
            const _Float16* brow = sB + (nt * 16 + m) * 136 + kb + hi * 8;
            bf[nt].f[0] = *(const float4*)(brow);
            bf[nt].f[1] = *(const float4*)(brow + 16);
        }
#pragma unroll
        for (int nt = 0; nt < 8; ++nt) {
            acc[nt] = __builtin_amdgcn_wmma_f32_16x16x32_f16(
                false, a.v, false, bf[nt].v, (short)0, acc[nt], false, false);
        }
    }

    // store: VGPR r, lanes0-15 -> M=r, lanes16-31 -> M=8+r; N = nt*16 + m
    // fused: S = C * dinv^2 (self-loop init), zero for padded rows
#pragma unroll
    for (int r = 0; r < 8; ++r) {
        const long long grow = rowBase + hi * 8 + r;
        const float dn = (grow < N) ? dinv[grow] : 0.0f;
        const float dn2 = dn * dn;
#pragma unroll
        for (int nt = 0; nt < 8; ++nt) {
            const float c = acc[nt][r];
            const long long off = grow * 128 + nt * 16 + m;
            C[off] = c;
            S[off] = c * dn2;
        }
    }
}

// ---------------- pooling + head ----------------

// one wave per node: atomic scatter-mean accumulation into (graph,cluster) segments
__global__ __launch_bounds__(256)
void k_pool(const float* __restrict__ S3, const float* __restrict__ b3,
            const float* __restrict__ x, float* __restrict__ sums,
            float* __restrict__ cnt, int N, int nodesPerGraph) {
    int i = blockIdx.x * 8 + (threadIdx.x >> 5);
    if (i >= N) return;
    int lane = threadIdx.x & 31;
    float c1 = x[(long long)i * 128 + 127];
    float c2 = x[(long long)i * 128 + 126];
    int cl = (int)(c1 + 2.0f * c2 + 0.5f);
    int cb = (i / nodesPerGraph) * 3 + cl;
    float4 v  = ((const float4*)(S3 + (long long)i * 128))[lane];
    float4 bb = ((const float4*)b3)[lane];
    float* out = sums + (long long)cb * 128 + lane * 4;
    atomicAdd(out + 0, v.x + bb.x);
    atomicAdd(out + 1, v.y + bb.y);
    atomicAdd(out + 2, v.z + bb.z);
    atomicAdd(out + 3, v.w + bb.w);
    if (lane == 0) atomicAdd(&cnt[cb], 1.0f);
}

// one wave per graph: z = (sums/cnt) . Wl + bl, sigmoid
__global__ __launch_bounds__(256)
void k_head(const float* __restrict__ sums, const float* __restrict__ cnt,
            const float* __restrict__ Wl, const float* __restrict__ bl,
            float* __restrict__ out, int B) {
    int b = blockIdx.x * 8 + (threadIdx.x >> 5);
    int lane = threadIdx.x & 31;
    float acc = 0.f;
    if (b < B) {
        for (int k = lane; k < 384; k += 32) {
            int seg = b * 3 + (k >> 7);
            int j = k & 127;
            float cn = fmaxf(cnt[seg], 1.0f);
            acc += (sums[seg * 128 + j] / cn) * Wl[k];
        }
    }
#pragma unroll
    for (int off = 16; off > 0; off >>= 1) acc += __shfl_down(acc, off, 32);
    if (b < B && lane == 0) out[b] = 1.0f / (1.0f + __expf(-(acc + bl[0])));
}

// ---------------- launch ----------------

extern "C" void kernel_launch(void* const* d_in, const int* in_sizes, int n_in,
                              void* d_out, int out_size, void* d_ws, size_t ws_size,
                              hipStream_t stream) {
    const float* x  = (const float*)d_in[0];
    const int*   ei = (const int*)d_in[1];
    const float* W1 = (const float*)d_in[3];
    const float* b1 = (const float*)d_in[4];
    const float* W2 = (const float*)d_in[5];
    const float* b2 = (const float*)d_in[6];
    const float* W3 = (const float*)d_in[7];
    const float* b3 = (const float*)d_in[8];
    const float* Wl = (const float*)d_in[9];
    const float* bl = (const float*)d_in[10];
    float* out = (float*)d_out;

    const int N = in_sizes[0] / 128;
    const int E = in_sizes[1] / 2;
    const int B = out_size;
    const int nodesPerGraph = N / B;
    const int Npad = ((N + 127) / 128) * 128;

    const int* srcI = ei;
    const int* dstI = ei + E;

    char* p = (char*)d_ws;
    _Float16* A16 = (_Float16*)p; p += (size_t)Npad * 128 * sizeof(_Float16);
    float*    hW  = (float*)p;    p += (size_t)Npad * 128 * sizeof(float);
    float*    S   = (float*)p;    p += (size_t)Npad * 128 * sizeof(float);
    _Float16* Wt  = (_Float16*)p; p += (size_t)128 * 128 * sizeof(_Float16);
    float*    dinv= (float*)p;    p += (size_t)N * sizeof(float);
    float*    sums= (float*)p;    p += (size_t)B * 3 * 128 * sizeof(float);
    float*    cnt = (float*)p;    p += (size_t)B * 3 * sizeof(float);

    const int TPB = 256;
    auto cdiv = [](long long a, long long b) { return (int)((a + b - 1) / b); };

    // degrees -> dinv
    k_fill<<<cdiv(N, TPB), TPB, 0, stream>>>(dinv, 1.0f, N);
    k_deg_edges<<<cdiv(E, TPB), TPB, 0, stream>>>(dstI, dinv, E);
    k_rsqrt_inplace<<<cdiv(N, TPB), TPB, 0, stream>>>(dinv, N);

    // zero pooling accumulators (sums and cnt are contiguous)
    k_fill<<<cdiv((long long)B * 3 * 129, TPB), TPB, 0, stream>>>(sums, 0.0f, B * 3 * 129);

    // layer input: f16 conversion of x (padded)
    const long long nV4 = (long long)Npad * 32;
    k_conv_in<<<cdiv(nV4, TPB), TPB, 0, stream>>>(x, A16, N, Npad);

    const float* Ws[3] = {W1, W2, W3};
    const float* bs[3] = {b1, b2, b3};
    const int gemmBlocks = Npad / 128;

    for (int l = 0; l < 3; ++l) {
        k_conv_w<<<64, TPB, 0, stream>>>(Ws[l], Wt);
        // GEMM writes hW and the self-loop-scaled S in one pass
        k_gemm_wmma<<<gemmBlocks, TPB, 0, stream>>>(A16, Wt, hW, S, dinv, N);
        k_edge_scatter<<<cdiv(E, 8), TPB, 0, stream>>>(srcI, dstI, dinv, hW, S, E);
        if (l < 2) {
            k_finish_relu<<<cdiv(nV4, TPB), TPB, 0, stream>>>(S, bs[l], A16, N, Npad);
        }
    }

    // pooling over (graph, cluster) segments; layer-3 bias folded in here
    k_pool<<<cdiv(N, 8), TPB, 0, stream>>>(S, b3, x, sums, cnt, N, nodesPerGraph);

    // head: 250 graphs, one wave each
    k_head<<<cdiv((long long)B * 32, TPB), TPB, 0, stream>>>(sums, cnt, Wl, bl, out, B);
}
